// VQ_37898791420621
// MI455X (gfx1250) — matrix operand order repaired and checked by reading, not compiled, not run on hardware
//
#include <hip/hip_runtime.h>

// ---------------------------------------------------------------------------
// VQ-VAE vector quantization for MI455X (gfx1250, wave32, WMMA).
//
// sim GEMM via V_WMMA_F32_16X16X32_BF16 with a bf16-hi/lo split of both
// operands (hi*hi + hi*lo + lo*hi) -> ~2^-16 relative error on sim, argmin-
// safe, at bf16 matrix throughput. Two independent accumulator chains (one
// per 32-deep K chunk) to double WMMA ILP. Kernel is HBM-bound:
// ~67MB mandatory traffic @ 23.3 TB/s ~= 2.9us; bf16x2 compute ~= 5us issue
// over the whole device, overlapped across 2 WGs/WGP.
// ---------------------------------------------------------------------------

typedef __attribute__((ext_vector_type(16))) __bf16 v16bf;
typedef __attribute__((ext_vector_type(8)))  float  v8f;

union V16 {
    v16bf v;
    unsigned short u[16];
    uint4 q[2];
};

__device__ __forceinline__ unsigned short bf16_rne(float f) {
    unsigned int u = __float_as_uint(f);
    unsigned int lsb = (u >> 16) & 1u;
    u += 0x7fffu + lsb;                 // round-to-nearest-even into top 16 bits
    return (unsigned short)(u >> 16);
}
__device__ __forceinline__ float bf16_f32(unsigned short h) {
    return __uint_as_float(((unsigned int)h) << 16);
}

#define EMB_D   64      // depth
#define EMB_K   512     // number of codes
#define WAVES   8       // waves per block (256 threads, wave32)
#define ROWS_PW 16      // rows per wave
#define ROWS_WG (WAVES * ROWS_PW)   // 128 rows per workgroup

// dynamic LDS layout (bytes)
#define LDS_EHI   0                         // 64*512 bf16 transposed: ehiT[k*64+d]
#define LDS_ELO   (LDS_EHI + EMB_D*EMB_K*2) // 64*512 bf16 lo part
#define LDS_SQE   (LDS_ELO + EMB_D*EMB_K*2) // 512 f32 ||e_k||^2
#define LDS_IDX   (LDS_SQE + EMB_K*4)       // 128 i32 winning code per row
#define LDS_WS    (LDS_IDX + ROWS_WG*4)     // 8 f32 per-wave loss partials
#define LDS_TOTAL (LDS_WS + WAVES*4)        // ~133.7 KB (2 WGs per 320KB WGP)

__global__ void vq_main_kernel(const float* __restrict__ x,
                               const float* __restrict__ embeds,
                               float* __restrict__ out,
                               float* __restrict__ partial) {
    extern __shared__ char smem[];
    unsigned short* EHI = (unsigned short*)(smem + LDS_EHI);
    unsigned short* ELO = (unsigned short*)(smem + LDS_ELO);
    float*          SQE = (float*)(smem + LDS_SQE);
    int*            IDX = (int*)(smem + LDS_IDX);
    float*          WS  = (float*)(smem + LDS_WS);

    const int t    = threadIdx.x;
    const int lane = t & 31;
    const int w    = t >> 5;
    const int bid  = blockIdx.x;

    // ------------------------------------------------------------------
    // Stage embeds into LDS: transposed bf16 hi/lo + exact fp32 sq-norms.
    // Thread t owns codes t and t+256 -> global reads coalesced across a
    // wave at each depth d; deterministic ||e||^2 accumulation order.
    // ------------------------------------------------------------------
    #pragma unroll
    for (int kk = 0; kk < 2; ++kk) {
        const int k = t + kk * 256;
        float sq = 0.f;
        for (int d = 0; d < EMB_D; ++d) {
            float v = embeds[d * EMB_K + k];
            sq += v * v;
            unsigned short hi = bf16_rne(v);
            EHI[k * EMB_D + d] = hi;
            ELO[k * EMB_D + d] = bf16_rne(v - bf16_f32(hi));
        }
        SQE[k] = sq;
    }
    __syncthreads();

    // ------------------------------------------------------------------
    // A fragment: 16 rows of x, contiguous K order (element s = k 16*half + s per chunk).
// lane<16  (row = lane):    K in {0-15,32-47}
// lane>=16 (row = lane-16): K in {16-31,48-63}
    // ------------------------------------------------------------------
    const int rowBase = bid * ROWS_WG + w * ROWS_PW;
    const int rowInW  = lane & 15;
    const size_t rowOff = (size_t)(rowBase + rowInW) * EMB_D;
    const float4* xr = (const float4*)(x + rowOff);
    const int b0 = (lane < 16) ? 0 : 4;

    float4 f4[8];
    #pragma unroll
    for (int j2 = 0; j2 < 8; ++j2)
        f4[j2] = xr[b0 + (j2 >> 2) * 8 + (j2 & 3)];

    V16 aHi[2], aLo[2];
    #pragma unroll
    for (int c = 0; c < 2; ++c) {
        const float* fp = (const float*)&f4[4 * c];   // 16 floats for chunk c
        #pragma unroll
        for (int i = 0; i < 16; ++i) {
            float v = fp[i];
            unsigned short hi = bf16_rne(v);
            aHi[c].u[i] = hi;
            aLo[c].u[i] = bf16_rne(v - bf16_f32(hi));
        }
    }

    // ------------------------------------------------------------------
    // 32 code tiles; per tile two INDEPENDENT 3-deep WMMA chains (one per
    // 32-wide K chunk), interleaved for matrix-pipe ILP. Streaming argmin
    // on  dist = ||e||^2 - 2*(acc0 + acc1).
    // ------------------------------------------------------------------
    float runMin[8];
    int   runIdx[8];
    #pragma unroll
    for (int v = 0; v < 8; ++v) { runMin[v] = 3.4e38f; runIdx[v] = 0; }

    const int half = lane >> 4;            // B layout: lanes 0-15 K-low, 16-31 K-high
    const int nsel = lane & 15;            // column within tile

    #pragma unroll 2
    for (int j = 0; j < EMB_K / 16; ++j) {
        const int col  = j * 16 + nsel;
        const int boff = col * EMB_D + 16 * half;   // ushort offset of K-chunk 0

        V16 bHi0, bLo0, bHi1, bLo1;
        {
            const uint4* ph0 = (const uint4*)(EHI + boff);
            const uint4* pl0 = (const uint4*)(ELO + boff);
            const uint4* ph1 = (const uint4*)(EHI + boff + 32);
            const uint4* pl1 = (const uint4*)(ELO + boff + 32);
            bHi0.q[0] = ph0[0]; bHi0.q[1] = ph0[1];
            bLo0.q[0] = pl0[0]; bLo0.q[1] = pl0[1];
            bHi1.q[0] = ph1[0]; bHi1.q[1] = ph1[1];
            bLo1.q[0] = pl1[0]; bLo1.q[1] = pl1[1];
        }
        const float se = SQE[col];

        v8f acc0 = {}, acc1 = {};
        acc0 = __builtin_amdgcn_wmma_f32_16x16x32_bf16(
            false, aHi[0].v, false, bHi0.v, (short)0, acc0, false, false);
        acc1 = __builtin_amdgcn_wmma_f32_16x16x32_bf16(
            false, aHi[1].v, false, bHi1.v, (short)0, acc1, false, false);
        acc0 = __builtin_amdgcn_wmma_f32_16x16x32_bf16(
            false, aHi[0].v, false, bLo0.v, (short)0, acc0, false, false);
        acc1 = __builtin_amdgcn_wmma_f32_16x16x32_bf16(
            false, aHi[1].v, false, bLo1.v, (short)0, acc1, false, false);
        acc0 = __builtin_amdgcn_wmma_f32_16x16x32_bf16(
            false, aLo[0].v, false, bHi0.v, (short)0, acc0, false, false);
        acc1 = __builtin_amdgcn_wmma_f32_16x16x32_bf16(
            false, aLo[1].v, false, bHi1.v, (short)0, acc1, false, false);

        #pragma unroll
        for (int v = 0; v < 8; ++v) {
            float dist = __builtin_fmaf(-2.0f, acc0[v],
                         __builtin_fmaf(-2.0f, acc1[v], se));
            bool take = dist < runMin[v];              // strict < keeps first min
            runIdx[v] = take ? col  : runIdx[v];
            runMin[v] = take ? dist : runMin[v];
        }
    }

    // Cross-lane argmin within each 16-lane half, branchless selects
    // (tie-break: lowest index, matching jnp.argmin).
    #pragma unroll
    for (int m = 1; m <= 8; m <<= 1) {
        #pragma unroll
        for (int v = 0; v < 8; ++v) {
            float ov = __shfl_xor(runMin[v], m, 32);
            int   oi = __shfl_xor(runIdx[v], m, 32);
            bool take = (ov < runMin[v]) | ((ov == runMin[v]) & (oi < runIdx[v]));
            runMin[v] = take ? ov : runMin[v];
            runIdx[v] = take ? oi : runIdx[v];
        }
    }
    // C layout: VGPR v holds row v (lanes 0-15) / row v+8 (lanes 16-31)
    if (nsel == 0) {
        #pragma unroll
        for (int v = 0; v < 8; ++v)
            IDX[w * ROWS_PW + half * 8 + v] = runIdx[v];
    }
    __syncthreads();

    // ------------------------------------------------------------------
    // Output: gather exact fp32 codebook columns (L2-resident), write
    // quantized_st (== quantized), accumulate sum((q-x)^2) deterministically.
    // Each lane handles depths {lane, lane+32} of each of the wave's 16 rows.
    // ------------------------------------------------------------------
    float lsum = 0.f;
    for (int r = 0; r < ROWS_PW; ++r) {
        const int k = IDX[w * ROWS_PW + r];
        const size_t o = (size_t)(rowBase + r) * EMB_D;
        const int d0 = lane, d1 = lane + 32;
        float q0 = embeds[d0 * EMB_K + k];
        float q1 = embeds[d1 * EMB_K + k];
        float x0 = x[o + d0];
        float x1 = x[o + d1];
        out[o + d0] = q0;
        out[o + d1] = q1;
        float e0 = q0 - x0, e1 = q1 - x1;
        lsum += e0 * e0 + e1 * e1;
    }
    #pragma unroll
    for (int m = 1; m <= 16; m <<= 1)
        lsum += __shfl_xor(lsum, m, 32);
    if (lane == 0) WS[w] = lsum;
    __syncthreads();
    if (t == 0) {
        float s = 0.f;
        #pragma unroll
        for (int i = 0; i < WAVES; ++i) s += WS[i];
        partial[bid] = s;
    }
}

// Deterministic final reduction + loss scale: loss = 1.25 * sum / count
__global__ void vq_finalize_kernel(const float* __restrict__ partial,
                                   float* __restrict__ loss,
                                   int n, float scale) {
    if (blockIdx.x == 0 && threadIdx.x == 0) {
        float s = 0.f;
        for (int i = 0; i < n; ++i) s += partial[i];
        *loss = scale * s;
    }
}

extern "C" void kernel_launch(void* const* d_in, const int* in_sizes, int n_in,
                              void* d_out, int out_size, void* d_ws, size_t ws_size,
                              hipStream_t stream) {
    const float* x      = (const float*)d_in[0];   // (32,64,64,64) f32
    const float* embeds = (const float*)d_in[1];   // (64,512) f32
    float* out     = (float*)d_out;                // [quantized_st (N*64) | loss (1)]
    float* partial = (float*)d_ws;                 // one f32 per workgroup

    const int nElems = in_sizes[0];                // 8,388,608
    const int nRows  = nElems / EMB_D;             // 131,072
    const int blocks = nRows / ROWS_WG;            // 1,024

    hipLaunchKernelGGL(vq_main_kernel, dim3(blocks), dim3(WAVES * 32),
                       LDS_TOTAL, stream, x, embeds, out, partial);

    const float scale = 1.25f / (float)nElems;     // BETA*c_loss + cb_loss = 1.25*mse
    hipLaunchKernelGGL(vq_finalize_kernel, dim3(1), dim3(32), 0, stream,
                       partial, out + nElems, blocks, scale);
}
